// DendriticAttentionLayer_14405320311231
// MI455X (gfx1250) — matrix operand order repaired and checked by reading, not compiled, not run on hardware
//
#include <hip/hip_runtime.h>
#include <hip/hip_bf16.h>

// ---------------------------------------------------------------------------
// Types for CDNA5 WMMA
// ---------------------------------------------------------------------------
typedef __attribute__((ext_vector_type(16))) __bf16 v16bf;
typedef __attribute__((ext_vector_type(8)))  float  v8f;

union FragAB {
    uint4 q[2];          // 2 x 16B = 32B
    v16bf v;             // 16 bf16 = 32B
};

// ---------------------------------------------------------------------------
// Helpers
// ---------------------------------------------------------------------------
__device__ __forceinline__ unsigned short f32_to_bf16(float f) {
    unsigned int u = __float_as_uint(f);
    unsigned int r = u + 0x7FFFu + ((u >> 16) & 1u);   // round to nearest even
    return (unsigned short)(r >> 16);
}

__device__ __forceinline__ void unpack8_bf16(uint4 p, float* f) {
    unsigned int w[4] = {p.x, p.y, p.z, p.w};
#pragma unroll
    for (int i = 0; i < 4; ++i) {
        f[2 * i + 0] = __uint_as_float((w[i] & 0x0000FFFFu) << 16);
        f[2 * i + 1] = __uint_as_float(w[i] & 0xFFFF0000u);
    }
}

// CDNA5 async DMA: global memory -> LDS, 16B per lane, tracked by ASYNCcnt.
__device__ __forceinline__ void async_load_b128(const void* gptr, void* sptr) {
    unsigned int lds = (unsigned int)(unsigned long long)(uintptr_t)sptr;
    unsigned long long ga = (unsigned long long)(uintptr_t)gptr;
    asm volatile("global_load_async_to_lds_b128 %0, %1, off"
                 :: "v"(lds), "v"(ga) : "memory");
}

__device__ __forceinline__ void wait_asynccnt0() {
    asm volatile("s_wait_asynccnt 0x0" ::: "memory");
}

// ---------------------------------------------------------------------------
// f32 -> bf16 conversion kernel (vectorized x4)
// ---------------------------------------------------------------------------
__global__ __launch_bounds__(256)
void convert_f32_to_bf16(const float* __restrict__ in,
                         unsigned short* __restrict__ out, int n4) {
    int i = blockIdx.x * blockDim.x + threadIdx.x;
    if (i >= n4) return;
    const float4 f = ((const float4*)in)[i];
    ushort4 o;
    o.x = f32_to_bf16(f.x);
    o.y = f32_to_bf16(f.y);
    o.z = f32_to_bf16(f.z);
    o.w = f32_to_bf16(f.w);
    ((ushort4*)out)[i] = o;
}

// ---------------------------------------------------------------------------
// WMMA GEMM:  out[m,n] = sum_k A[m,k] * W[n,k] + bias[n]      (A: MxK, W: NxK)
// mode 0: store bf16 activation
// mode 1: v_out blend + spike epilogue, write two f32 outputs
// Block: 256 threads (8 wave32), 128x128 tile; wave tile 32x64 (2x4 of 16x16)
// K staged 64 deep, double-buffered LDS filled by async global->LDS DMA.
// ---------------------------------------------------------------------------
#define TILE_K 64
#define KPAD   72        // 64 + 8 bf16 pad: keeps 16B alignment, kills bank conflicts

__global__ __launch_bounds__(256)
void gemm_wmma(const unsigned short* __restrict__ A,
               const unsigned short* __restrict__ W,
               const float* __restrict__ bias,
               unsigned short* __restrict__ out16,
               const float* __restrict__ v_in,
               float* __restrict__ spike_out,
               float* __restrict__ vfin_out,
               int mode) {
    const int Nn = 4096;
    const int K  = 4096;

    __shared__ unsigned short As[2][128 * KPAD];
    __shared__ unsigned short Bs[2][128 * KPAD];

    const int tid  = threadIdx.x;
    const int lane = tid & 31;
    const int wave = tid >> 5;
    const int wm   = wave >> 1;     // 0..3  (M direction, 32 rows each)
    const int wn   = wave & 1;      // 0..1  (N direction, 64 cols each)
    const int half = lane >> 4;     // 0/1
    const int n16  = lane & 15;

    const int blockM = blockIdx.y * 128;
    const int blockN = blockIdx.x * 128;

    const v8f zacc = {0.f, 0.f, 0.f, 0.f, 0.f, 0.f, 0.f, 0.f};
    v8f acc[2][4];
#pragma unroll
    for (int ti = 0; ti < 2; ++ti)
#pragma unroll
        for (int tj = 0; tj < 4; ++tj) acc[ti][tj] = zacc;

    // cooperative stage: thread t owns row t>>1, 64B half-row (t&1)
    const int lrow = tid >> 1;
    const int lseg = tid & 1;                    // 0/1 -> k sub-range of 32
    const unsigned short* gA = A + (size_t)(blockM + lrow) * K + lseg * 32;
    const unsigned short* gB = W + (size_t)(blockN + lrow) * K + lseg * 32;
    const int srow = lrow * KPAD + lseg * 32;

    // prologue: stage k0 = 0 into buffer 0
#pragma unroll
    for (int c = 0; c < 4; ++c) {
        async_load_b128(gA + c * 8, &As[0][srow + c * 8]);
        async_load_b128(gB + c * 8, &Bs[0][srow + c * 8]);
    }
    wait_asynccnt0();
    __syncthreads();

    int cur = 0;
    for (int k0 = 0; k0 < K; k0 += TILE_K) {
        // stage next tile into the other buffer while we compute
        if (k0 + TILE_K < K) {
            const int nxt = cur ^ 1;
#pragma unroll
            for (int c = 0; c < 4; ++c) {
                async_load_b128(gA + k0 + TILE_K + c * 8, &As[nxt][srow + c * 8]);
                async_load_b128(gB + k0 + TILE_K + c * 8, &Bs[nxt][srow + c * 8]);
            }
        }

        // compute on current buffer: two K=32 WMMA steps
#pragma unroll
        for (int kk = 0; kk < TILE_K; kk += 32) {
            // B fragments: 32x16 bf16, lane n16 = col, elems 0..7 -> K = kk + half*8 + 0..7, elems 8..15 -> K = kk + 16 + half*8 + 0..7
            FragAB fb[4];
#pragma unroll
            for (int tj = 0; tj < 4; ++tj) {
                const unsigned short* bp =
                    &Bs[cur][(wn * 64 + tj * 16 + n16) * KPAD + kk + half * 8];
                fb[tj].q[0] = *(const uint4*)(bp);
                fb[tj].q[1] = *(const uint4*)(bp + 16);
            }
            // A fragments: 16x32 bf16, lane n16 = row,
            //   elems 0..7  -> K = kk + half*8 + 0..7
            //   elems 8..15 -> K = kk + 16 + half*8 + 0..7
#pragma unroll
            for (int ti = 0; ti < 2; ++ti) {
                FragAB fa;
                const unsigned short* ap =
                    &As[cur][(wm * 32 + ti * 16 + n16) * KPAD + kk + half * 8];
                fa.q[0] = *(const uint4*)(ap);
                fa.q[1] = *(const uint4*)(ap + 16);
#pragma unroll
                for (int tj = 0; tj < 4; ++tj) {
                    acc[ti][tj] = __builtin_amdgcn_wmma_f32_16x16x32_bf16(
                        false, fa.v, false, fb[tj].v, (short)0, acc[ti][tj],
                        false, false);
                }
            }
        }

        // my DMA landed; barrier makes everyone's DMA + LDS reads visible
        wait_asynccnt0();
        __syncthreads();
        cur ^= 1;
    }

    // beta average for the final-stage LIF blend (compile-time foldable)
    float ba = 0.f;
#pragma unroll
    for (int h = 0; h < 16; ++h)
        ba += 1.f - 1.f / (2.f + (float)h * (14.f / 15.f));
    ba *= (1.f / 16.f);

    // Epilogue: C/D layout: lane n16 = col, vgpr r -> row half*8 + r
#pragma unroll
    for (int ti = 0; ti < 2; ++ti) {
#pragma unroll
        for (int tj = 0; tj < 4; ++tj) {
            const int col = blockN + wn * 64 + tj * 16 + n16;
            const float bcol = bias[col];
#pragma unroll
            for (int r = 0; r < 8; ++r) {
                const int row = blockM + wm * 32 + ti * 16 + half * 8 + r;
                const size_t idx = (size_t)row * Nn + col;
                const float val = acc[ti][tj][r] + bcol;
                if (mode == 0) {
                    out16[idx] = f32_to_bf16(val);
                } else {
                    const float vn = ba * v_in[idx] + (1.f - ba) * val;
                    const float sp = (vn - 1.f) > 0.f ? 1.f : 0.f;
                    spike_out[idx] = sp;
                    vfin_out[idx]  = vn - sp;    // V_TH = 1.0
                }
            }
        }
    }
}

// ---------------------------------------------------------------------------
// Middle kernel: per-row head blend + softmax + gate.
// One block (256 threads) per batch row; 16 threads per head (D=256).
// ---------------------------------------------------------------------------
__global__ __launch_bounds__(256)
void middle_kernel(const unsigned short* __restrict__ qb,
                   const unsigned short* __restrict__ kb,
                   const unsigned short* __restrict__ vb,
                   const unsigned short* __restrict__ gb,
                   const float* __restrict__ v_heads,
                   const float* __restrict__ temperature,
                   unsigned short* __restrict__ pre) {
    const int b = blockIdx.x;
    const int t = threadIdx.x;
    __shared__ float partial[256];
    __shared__ float logits[16];

    const int h = t >> 4;                              // head of this thread
    const float tau  = 2.f + (float)h * (14.f / 15.f); // linspace(2,16,16)
    const float beta = 1.f - 1.f / tau;

    const size_t base = (size_t)b * 4096 + (size_t)t * 16;

    float qv[16], kv[16], vv[16], gv[16], vh[16];
    {
        const uint4* qp = (const uint4*)(qb + base);
        const uint4* kp = (const uint4*)(kb + base);
        const uint4* vp = (const uint4*)(vb + base);
        const uint4* gp = (const uint4*)(gb + base);
        unpack8_bf16(qp[0], qv);     unpack8_bf16(qp[1], qv + 8);
        unpack8_bf16(kp[0], kv);     unpack8_bf16(kp[1], kv + 8);
        unpack8_bf16(vp[0], vv);     unpack8_bf16(vp[1], vv + 8);
        unpack8_bf16(gp[0], gv);     unpack8_bf16(gp[1], gv + 8);
        const float4* hp = (const float4*)(v_heads + base);
#pragma unroll
        for (int i = 0; i < 4; ++i) {
            float4 f = hp[i];
            vh[4 * i + 0] = f.x; vh[4 * i + 1] = f.y;
            vh[4 * i + 2] = f.z; vh[4 * i + 3] = f.w;
        }
    }

    float s = 0.f;
#pragma unroll
    for (int j = 0; j < 16; ++j) {
        const float vhn = beta * vh[j] + (1.f - beta) * qv[j] * kv[j];
        s += vhn;
    }
    partial[t] = s;
    __syncthreads();

    if (t < 16) {
        float l = 0.f;
#pragma unroll
        for (int i = 0; i < 16; ++i) l += partial[t * 16 + i];
        logits[t] = l * (1.f / 16.f) / temperature[0];  // / sqrt(D)=16
    }
    __syncthreads();

    float mx = logits[0];
#pragma unroll
    for (int i = 1; i < 16; ++i) mx = fmaxf(mx, logits[i]);
    float denom = 0.f;
#pragma unroll
    for (int i = 0; i < 16; ++i) denom += __expf(logits[i] - mx);
    const float attn = __expf(logits[h] - mx) / denom;

    unsigned short o[16];
#pragma unroll
    for (int j = 0; j < 16; ++j) {
        const float gate = 1.f / (1.f + __expf(-gv[j]));
        o[j] = f32_to_bf16(attn * vv[j] * gate);
    }
    uint4* op = (uint4*)(pre + base);
    uint4 w0, w1;
    w0.x = (unsigned int)o[0]  | ((unsigned int)o[1]  << 16);
    w0.y = (unsigned int)o[2]  | ((unsigned int)o[3]  << 16);
    w0.z = (unsigned int)o[4]  | ((unsigned int)o[5]  << 16);
    w0.w = (unsigned int)o[6]  | ((unsigned int)o[7]  << 16);
    w1.x = (unsigned int)o[8]  | ((unsigned int)o[9]  << 16);
    w1.y = (unsigned int)o[10] | ((unsigned int)o[11] << 16);
    w1.z = (unsigned int)o[12] | ((unsigned int)o[13] << 16);
    w1.w = (unsigned int)o[14] | ((unsigned int)o[15] << 16);
    op[0] = w0;
    op[1] = w1;
}

// ---------------------------------------------------------------------------
// Host launcher
// ---------------------------------------------------------------------------
extern "C" void kernel_launch(void* const* d_in, const int* in_sizes, int n_in,
                              void* d_out, int out_size, void* d_ws,
                              size_t ws_size, hipStream_t stream) {
    (void)in_sizes; (void)n_in; (void)out_size; (void)ws_size;

    const float* x    = (const float*)d_in[0];
    const float* Wq   = (const float*)d_in[1];
    const float* bq   = (const float*)d_in[2];
    const float* Wk   = (const float*)d_in[3];
    const float* bk   = (const float*)d_in[4];
    const float* Wv   = (const float*)d_in[5];
    const float* bv   = (const float*)d_in[6];
    const float* Wg   = (const float*)d_in[7];
    const float* bg   = (const float*)d_in[8];
    const float* Wo   = (const float*)d_in[9];
    const float* bo   = (const float*)d_in[10];
    const float* temp = (const float*)d_in[11];
    const float* v_heads = (const float*)d_in[12];
    const float* v_out   = (const float*)d_in[13];

    const size_t BC = (size_t)4096 * 4096;   // 16,777,216
    float* spike = (float*)d_out;
    float* vfin  = (float*)d_out + BC;

    unsigned short* xb  = (unsigned short*)d_ws;
    unsigned short* wqb = xb  + BC;
    unsigned short* wkb = wqb + BC;
    unsigned short* wvb = wkb + BC;
    unsigned short* wgb = wvb + BC;
    unsigned short* wob = wgb + BC;
    unsigned short* qb  = wob + BC;
    unsigned short* kb  = qb  + BC;
    unsigned short* vb2 = kb  + BC;
    unsigned short* gb2 = vb2 + BC;
    unsigned short* pre = gb2 + BC;

    // 1) convert x and weights to bf16
    const int n4 = (int)(BC / 4);
    const int cblocks = (n4 + 255) / 256;
    convert_f32_to_bf16<<<cblocks, 256, 0, stream>>>(x,  xb,  n4);
    convert_f32_to_bf16<<<cblocks, 256, 0, stream>>>(Wq, wqb, n4);
    convert_f32_to_bf16<<<cblocks, 256, 0, stream>>>(Wk, wkb, n4);
    convert_f32_to_bf16<<<cblocks, 256, 0, stream>>>(Wv, wvb, n4);
    convert_f32_to_bf16<<<cblocks, 256, 0, stream>>>(Wg, wgb, n4);
    convert_f32_to_bf16<<<cblocks, 256, 0, stream>>>(Wo, wob, n4);

    // 2) q,k,v,gate projections (bf16 WMMA, f32 accumulate, bf16 activations)
    dim3 ggrid(32, 32, 1);
    gemm_wmma<<<ggrid, 256, 0, stream>>>(xb, wqb, bq, qb,  nullptr, nullptr, nullptr, 0);
    gemm_wmma<<<ggrid, 256, 0, stream>>>(xb, wkb, bk, kb,  nullptr, nullptr, nullptr, 0);
    gemm_wmma<<<ggrid, 256, 0, stream>>>(xb, wvb, bv, vb2, nullptr, nullptr, nullptr, 0);
    gemm_wmma<<<ggrid, 256, 0, stream>>>(xb, wgb, bg, gb2, nullptr, nullptr, nullptr, 0);

    // 3) head blend + softmax + gate -> pre-projection activation
    middle_kernel<<<4096, 256, 0, stream>>>(qb, kb, vb2, gb2, v_heads, temp, pre);

    // 4) output projection + LIF blend + spike epilogue
    gemm_wmma<<<ggrid, 256, 0, stream>>>(pre, wob, bo, nullptr, v_out, spike, vfin, 1);
}
